// RegionProposalNetwork_58342835749701
// MI455X (gfx1250) — matrix operand order repaired and checked by reading, not compiled, not run on hardware
//
#include <hip/hip_runtime.h>
#include <hip/hip_bf16.h>

// ---------------------------------------------------------------------------
// RPN for MI455X (gfx1250, wave32).
//  - 3x3 conv as implicit GEMM  M=4096,N=512,K=4608  via v_wmma_f32_16x16x32_f16
//  - 1x1 convs fused as GEMM    M=4096,N=64, K=512   via v_wmma_f32_16x16x32_f16
//  - decode / softmax / bitonic top-k / greedy NMS tail
// Fragment layouts follow CDNA5 ISA §7.12.2 exactly (A: two b128
// chunks per lane with K = 8*(lane>>4)+{0..7,16..23}; B: same two b128 chunks as A
// with K = 8*(lane>>4)+{0..7,16..23}; C/D: m = 8*(lane>>4)+vgpr, n = lane&15).
// Prefetches use locality 3 so they lower to WGP-scope global_prefetch
// (SYS-scope prefetch would only fill GL2, not the WGP cache we re-read from).
// ---------------------------------------------------------------------------

typedef __attribute__((ext_vector_type(16))) _Float16 v16h;
typedef __attribute__((ext_vector_type(8)))  _Float16 v8h;
typedef __attribute__((ext_vector_type(8)))  float    v8f;

union Frag16 { v16h v; v8h h[2]; };

#define HW        64
#define SPATIAL   4096          // 64*64
#define CIN       512
#define KTOT      4608          // 512*9
#define NANCH     36864         // 4096*9
#define PREK      6000
#define POSTK     300
#define SORT_N    65536
#define MIN_SZ    0.016f
#define NMS_TH    0.7f

// ---- output layout (f32 words) ----
#define OUT_LOCS   0            // 36864*4
#define OUT_SCORES 147456       // 36864*2
#define OUT_ROIS   221184       // 300*4
#define OUT_RIDX   222384       // 300

// ---- workspace layout (bytes) ----
constexpr size_t OFF_XP   = 0;                          // padded input HWC f16 [66][66][512]
constexpr size_t SZ_XP    = (size_t)66*66*512*2;
constexpr size_t OFF_W2   = OFF_XP + SZ_XP;             // conv1 weights f16 [512][4608]
constexpr size_t SZ_W2    = (size_t)512*4608*2;
constexpr size_t OFF_H    = OFF_W2 + SZ_W2;             // hidden f16 [4096][512]
constexpr size_t SZ_H     = (size_t)4096*512*2;
constexpr size_t OFF_W3   = OFF_H + SZ_H;               // 1x1 weights f16 [64][512]
constexpr size_t SZ_W3    = (size_t)64*512*2;
constexpr size_t OFF_B3   = OFF_W3 + SZ_W3;             // 1x1 bias f32 [64]
constexpr size_t SZ_B3    = 256;
constexpr size_t OFF_O2   = OFF_B3 + SZ_B3;             // 1x1 out f32 [4096][64]
constexpr size_t SZ_O2    = (size_t)4096*64*4;
constexpr size_t OFF_BOX  = OFF_O2 + SZ_O2;             // decoded boxes f32 [36864][4]
constexpr size_t SZ_BOX   = (size_t)NANCH*16;
constexpr size_t OFF_KEY  = OFF_BOX + SZ_BOX;           // sort keys f32 [65536]
constexpr size_t SZ_KEY   = (size_t)SORT_N*4;
constexpr size_t OFF_VAL  = OFF_KEY + SZ_KEY;           // sort vals i32 [65536]
constexpr size_t SZ_VAL   = (size_t)SORT_N*4;
constexpr size_t OFF_BSEL = OFF_VAL + SZ_VAL;           // top-6000 boxes f32 [6000][4]
constexpr size_t SZ_BSEL  = 96256;
constexpr size_t OFF_VLD  = OFF_BSEL + SZ_BSEL;         // valid flags i32 [6000]
constexpr size_t SZ_VLD   = 24064;
constexpr size_t OFF_KEEP = OFF_VLD + SZ_VLD;           // keep flags i32 [6000]

// ---------------------------------------------------------------------------
// Prep: f32 NCHW -> padded f16 HWC
__global__ void prep_xp(const float* __restrict__ x, _Float16* __restrict__ Xp) {
  int idx = blockIdx.x * blockDim.x + threadIdx.x;
  if (idx >= 66*66*512) return;
  int c  = idx & 511;
  int sp = idx >> 9;
  int cc = sp % 66;
  int r  = sp / 66;
  float v = 0.f;
  if (r >= 1 && r <= 64 && cc >= 1 && cc <= 64)
    v = x[(size_t)c*SPATIAL + (r-1)*HW + (cc-1)];
  Xp[idx] = (_Float16)v;
}

// conv1_w [O][C][3][3] f32 -> W2 [O][(ky*3+kx)*512+c] f16  (B^T layout)
__global__ void prep_w2(const float* __restrict__ w, _Float16* __restrict__ W2) {
  int idx = blockIdx.x * blockDim.x + threadIdx.x;
  if (idx >= 512*KTOT) return;
  int o = idx / KTOT;
  int k = idx - o*KTOT;
  int win = k >> 9;
  int c   = k & 511;
  int ky  = win / 3, kx = win - ky*3;
  W2[idx] = (_Float16)w[((size_t)o*512 + c)*9 + ky*3 + kx];
}

// loc_w[36][512] + score_w[18][512] -> W3[64][512] f16 (rows 54..63 zero), bias
__global__ void prep_w3(const float* __restrict__ loc_w, const float* __restrict__ score_w,
                        const float* __restrict__ loc_b, const float* __restrict__ score_b,
                        _Float16* __restrict__ W3, float* __restrict__ B3) {
  int idx = blockIdx.x * blockDim.x + threadIdx.x;
  if (idx >= 64*512) return;
  int n = idx >> 9;
  int k = idx & 511;
  float v = 0.f;
  if      (n < 36) v = loc_w[n*512 + k];
  else if (n < 54) v = score_w[(n-36)*512 + k];
  W3[idx] = (_Float16)v;
  if (k == 0) {
    float b = 0.f;
    if      (n < 36) b = loc_b[n];
    else if (n < 54) b = score_b[n-36];
    B3[n] = b;
  }
}

// ---------------------------------------------------------------------------
// GEMM1: H[m][n] = relu(sum_k A[m,k]*W2[n,k] + bias[n]); M=4096 N=512 K=4608
// block = 8 waves = 128x128 tile; wave = 64x32 (4x2 WMMA tiles)
__global__ __launch_bounds__(256) void gemm1_wmma(
    const _Float16* __restrict__ Xp, const _Float16* __restrict__ W2,
    const float* __restrict__ bias, _Float16* __restrict__ Hout) {
  const int lane = threadIdx.x & 31;
  const int wid  = threadIdx.x >> 5;
  const int wm   = wid >> 2;                 // 0..1 (rows of 64)
  const int wn   = wid & 3;                  // 0..3 (cols of 32)
  const int blkM = blockIdx.x * 128;
  const int blkN = blockIdx.y * 128;
  const int g    = lane >> 4;
  const int l15  = lane & 15;

  v8f acc[4][2] = {};

  for (int t = 0; t < KTOT/32; ++t) {
    const int kk  = t * 32;
    const int win = kk >> 9;                 // K-blocks never straddle windows
    const int ky  = win / 3, kx = win - ky*3;
    const int c0  = kk & 511;

    Frag16 a[4];
#pragma unroll
    for (int mi = 0; mi < 4; ++mi) {
      int m = blkM + wm*64 + mi*16 + l15;    // A: lanes 0-15 & 16-31 both M=0..15
      int y = m >> 6, x = m & 63;
      const _Float16* p = Xp + ((size_t)((y+ky)*66 + (x+kx)))*512 + c0 + 8*g;
      a[mi].h[0] = *(const v8h*)p;           // K = 8g + 0..7
      a[mi].h[1] = *(const v8h*)(p + 16);    // K = 8g + 16..23
      __builtin_prefetch(p + 32, 0, 3);      // WGP-scope prefetch of next K-block
    }
    Frag16 b[2];
#pragma unroll
    for (int ni = 0; ni < 2; ++ni) {
      int n = blkN + wn*32 + ni*16 + l15;
      const _Float16* p = W2 + (size_t)n*KTOT + kk + 8*g;
      b[ni].h[0] = *(const v8h*)p;           // K = 8g + 0..7
      b[ni].h[1] = *(const v8h*)(p + 16);    // K = 8g + 16..23
      __builtin_prefetch(p + 32, 0, 3);
    }
#pragma unroll
    for (int mi = 0; mi < 4; ++mi)
#pragma unroll
      for (int ni = 0; ni < 2; ++ni)
        acc[mi][ni] = __builtin_amdgcn_wmma_f32_16x16x32_f16(
            false, a[mi].v, false, b[ni].v, (short)0, acc[mi][ni], false, false);
  }

#pragma unroll
  for (int mi = 0; mi < 4; ++mi) {
#pragma unroll
    for (int ni = 0; ni < 2; ++ni) {
      int n = blkN + wn*32 + ni*16 + l15;
      float bn = bias[n];
#pragma unroll
      for (int v = 0; v < 8; ++v) {
        int m = blkM + wm*64 + mi*16 + 8*g + v;
        float val = acc[mi][ni][v] + bn;
        val = val > 0.f ? val : 0.f;
        Hout[(size_t)m*512 + n] = (_Float16)val;
      }
    }
  }
}

// GEMM2: O2[m][n] = sum_k H[m,k]*W3[n,k] + b3[n]; M=4096 N=64 K=512
// block = 4 waves = 128x64 tile; wave = 32x64 (2x4 WMMA tiles)
__global__ __launch_bounds__(128) void gemm2_wmma(
    const _Float16* __restrict__ Hin, const _Float16* __restrict__ W3,
    const float* __restrict__ b3, float* __restrict__ O2) {
  const int lane = threadIdx.x & 31;
  const int wid  = threadIdx.x >> 5;         // 0..3
  const int blkM = blockIdx.x * 128;
  const int g    = lane >> 4;
  const int l15  = lane & 15;

  v8f acc[2][4] = {};

  for (int t = 0; t < 512/32; ++t) {
    const int kk = t * 32;
    Frag16 a[2];
#pragma unroll
    for (int mi = 0; mi < 2; ++mi) {
      int m = blkM + wid*32 + mi*16 + l15;
      const _Float16* p = Hin + (size_t)m*512 + kk + 8*g;
      a[mi].h[0] = *(const v8h*)p;
      a[mi].h[1] = *(const v8h*)(p + 16);
      __builtin_prefetch(p + 32, 0, 3);
    }
    Frag16 b[4];
#pragma unroll
    for (int ni = 0; ni < 4; ++ni) {
      int n = ni*16 + l15;
      const _Float16* p = W3 + (size_t)n*512 + kk + 8*g;
      b[ni].h[0] = *(const v8h*)p;
      b[ni].h[1] = *(const v8h*)(p + 16);
    }
#pragma unroll
    for (int mi = 0; mi < 2; ++mi)
#pragma unroll
      for (int ni = 0; ni < 4; ++ni)
        acc[mi][ni] = __builtin_amdgcn_wmma_f32_16x16x32_f16(
            false, a[mi].v, false, b[ni].v, (short)0, acc[mi][ni], false, false);
  }

#pragma unroll
  for (int mi = 0; mi < 2; ++mi)
#pragma unroll
    for (int ni = 0; ni < 4; ++ni) {
      int n = ni*16 + l15;
      float bn = b3[n];
#pragma unroll
      for (int v = 0; v < 8; ++v) {
        int m = blkM + wid*32 + mi*16 + 8*g + v;
        O2[(size_t)m*64 + n] = acc[mi][ni][v] + bn;
      }
    }
}

// ---------------------------------------------------------------------------
// decode anchors, write rpn_locs / rpn_scores, softmax fg, min-size filter
__global__ void postproc(const float* __restrict__ O2, const float* __restrict__ anchors,
                         float* __restrict__ out_locs, float* __restrict__ out_scores,
                         float* __restrict__ boxes, float* __restrict__ keys,
                         int* __restrict__ vals) {
  int i = blockIdx.x * blockDim.x + threadIdx.x;
  if (i >= NANCH) return;
  int p = i / 9, a = i - p*9;
  const float* row = O2 + (size_t)p*64;
  float L0 = row[a*4+0], L1 = row[a*4+1], L2 = row[a*4+2], L3 = row[a*4+3];
  out_locs[i*4+0] = L0; out_locs[i*4+1] = L1;
  out_locs[i*4+2] = L2; out_locs[i*4+3] = L3;
  float s0 = row[36 + a*2], s1 = row[36 + a*2 + 1];
  out_scores[i*2+0] = s0; out_scores[i*2+1] = s1;
  float fg = 1.f / (1.f + __expf(s0 - s1));

  const float* an = anchors + (size_t)i*4;
  float aw = an[2]-an[0], ah = an[3]-an[1];
  float acx = an[0] + 0.5f*aw, acy = an[1] + 0.5f*ah;
  float cx = acx + L0*aw, cy = acy + L1*ah;
  float w  = aw * __expf(L2), h = ah * __expf(L3);
  float b0 = fminf(fmaxf(cx - 0.5f*w, 0.f), 1.f);
  float b1 = fminf(fmaxf(cy - 0.5f*h, 0.f), 1.f);
  float b2 = fminf(fmaxf(cx + 0.5f*w, 0.f), 1.f);
  float b3 = fminf(fmaxf(cy + 0.5f*h, 0.f), 1.f);
  boxes[i*4+0]=b0; boxes[i*4+1]=b1; boxes[i*4+2]=b2; boxes[i*4+3]=b3;
  bool ok = (b2-b0) >= MIN_SZ && (b3-b1) >= MIN_SZ;
  keys[i] = ok ? fg : -1e9f;
  vals[i] = i;
}

__global__ void pad_tail(float* __restrict__ keys, int* __restrict__ vals) {
  int i = NANCH + blockIdx.x * blockDim.x + threadIdx.x;
  if (i < SORT_N) { keys[i] = -3.0e38f; vals[i] = i; }
}

// descending bitonic compare-exchange step
__global__ void bitonic_step(float* __restrict__ keys, int* __restrict__ vals, int j, int k) {
  int i = blockIdx.x * blockDim.x + threadIdx.x;
  int ixj = i ^ j;
  if (ixj > i) {
    float ki = keys[i], kj = keys[ixj];
    bool up = (i & k) == 0;
    bool sw = up ? (ki < kj) : (ki > kj);
    if (sw) {
      keys[i] = kj; keys[ixj] = ki;
      int t = vals[i]; vals[i] = vals[ixj]; vals[ixj] = t;
    }
  }
}

__global__ void gather_top(const float* __restrict__ keys, const int* __restrict__ vals,
                           const float* __restrict__ boxes, float* __restrict__ bsel,
                           int* __restrict__ valid) {
  int r = blockIdx.x * blockDim.x + threadIdx.x;
  if (r >= PREK) return;
  int idx = vals[r];
  bsel[r*4+0] = boxes[idx*4+0];
  bsel[r*4+1] = boxes[idx*4+1];
  bsel[r*4+2] = boxes[idx*4+2];
  bsel[r*4+3] = boxes[idx*4+3];
  valid[r] = keys[r] > -1e8f ? 1 : 0;
}

// greedy sequential NMS in one workgroup; suppression bitmask in LDS
__global__ __launch_bounds__(1024) void nms_kernel(
    const float* __restrict__ bsel, const int* __restrict__ valid,
    int* __restrict__ keep, float* __restrict__ out) {
  __shared__ unsigned sup[(PREK + 31) / 32];
  __shared__ float bx[4];
  __shared__ int live;
  const int tid = threadIdx.x;

  for (int w = tid; w < (PREK + 31) / 32; w += 1024) sup[w] = 0u;
  for (int w = tid; w < 1500; w += 1024) out[OUT_ROIS + w] = 0.f;  // rois + roi_indices
  __syncthreads();

  for (int i = 0; i < PREK; ++i) {
    if (tid == 0) {
      bool s = (sup[i >> 5] >> (i & 31)) & 1u;
      int lv = (!s && valid[i]) ? 1 : 0;
      keep[i] = lv; live = lv;
      bx[0] = bsel[i*4+0]; bx[1] = bsel[i*4+1];
      bx[2] = bsel[i*4+2]; bx[3] = bsel[i*4+3];
    }
    __syncthreads();
    if (live) {
      float x0 = bx[0], y0 = bx[1], x1 = bx[2], y1 = bx[3];
      float ai = (x1 - x0) * (y1 - y0);
      for (int j = i + 1 + tid; j < PREK; j += 1024) {
        float u0 = bsel[j*4+0], v0 = bsel[j*4+1], u1 = bsel[j*4+2], v1 = bsel[j*4+3];
        float aj = (u1 - u0) * (v1 - v0);
        float lx = fmaxf(x0, u0), ly = fmaxf(y0, v0);
        float rx = fminf(x1, u1), ry = fminf(y1, v1);
        float iw = fmaxf(rx - lx, 0.f), ih = fmaxf(ry - ly, 0.f);
        float inter = iw * ih;
        float iou = inter / (ai + aj - inter + 1e-9f);
        if (iou > NMS_TH) atomicOr(&sup[j >> 5], 1u << (j & 31));
      }
    }
    __syncthreads();
  }

  if (tid == 0) {
    int rank = 0;
    for (int i = 0; i < PREK && rank < POSTK; ++i) {
      if (keep[i]) {
        out[OUT_ROIS + rank*4 + 0] = bsel[i*4+0];
        out[OUT_ROIS + rank*4 + 1] = bsel[i*4+1];
        out[OUT_ROIS + rank*4 + 2] = bsel[i*4+2];
        out[OUT_ROIS + rank*4 + 3] = bsel[i*4+3];
        ++rank;
      }
    }
  }
}

// ---------------------------------------------------------------------------
extern "C" void kernel_launch(void* const* d_in, const int* in_sizes, int n_in,
                              void* d_out, int out_size, void* d_ws, size_t ws_size,
                              hipStream_t stream) {
  (void)in_sizes; (void)n_in; (void)out_size; (void)ws_size;
  const float* x       = (const float*)d_in[0];
  const float* conv1_w = (const float*)d_in[1];
  const float* conv1_b = (const float*)d_in[2];
  const float* score_w = (const float*)d_in[3];
  const float* score_b = (const float*)d_in[4];
  const float* loc_w   = (const float*)d_in[5];
  const float* loc_b   = (const float*)d_in[6];
  const float* anchors = (const float*)d_in[7];
  float* out = (float*)d_out;

  char* ws = (char*)d_ws;
  _Float16* Xp   = (_Float16*)(ws + OFF_XP);
  _Float16* W2   = (_Float16*)(ws + OFF_W2);
  _Float16* Hbuf = (_Float16*)(ws + OFF_H);
  _Float16* W3   = (_Float16*)(ws + OFF_W3);
  float*    B3   = (float*)(ws + OFF_B3);
  float*    O2   = (float*)(ws + OFF_O2);
  float*    Box  = (float*)(ws + OFF_BOX);
  float*    Keys = (float*)(ws + OFF_KEY);
  int*      Vals = (int*)(ws + OFF_VAL);
  float*    Bsel = (float*)(ws + OFF_BSEL);
  int*      Vld  = (int*)(ws + OFF_VLD);
  int*      Keep = (int*)(ws + OFF_KEEP);

  prep_xp <<<(66*66*512 + 255)/256, 256, 0, stream>>>(x, Xp);
  prep_w2 <<<(512*KTOT + 255)/256, 256, 0, stream>>>(conv1_w, W2);
  prep_w3 <<<(64*512 + 255)/256, 256, 0, stream>>>(loc_w, score_w, loc_b, score_b, W3, B3);

  gemm1_wmma<<<dim3(SPATIAL/128, 512/128), 256, 0, stream>>>(Xp, W2, conv1_b, Hbuf);
  gemm2_wmma<<<SPATIAL/128, 128, 0, stream>>>(Hbuf, W3, B3, O2);

  postproc<<<(NANCH + 255)/256, 256, 0, stream>>>(O2, anchors,
      out + OUT_LOCS, out + OUT_SCORES, Box, Keys, Vals);
  pad_tail<<<(SORT_N - NANCH + 255)/256, 256, 0, stream>>>(Keys, Vals);

  for (int k = 2; k <= SORT_N; k <<= 1)
    for (int j = k >> 1; j > 0; j >>= 1)
      bitonic_step<<<SORT_N/256, 256, 0, stream>>>(Keys, Vals, j, k);

  gather_top<<<(PREK + 255)/256, 256, 0, stream>>>(Keys, Vals, Box, Bsel, Vld);
  nms_kernel<<<1, 1024, 0, stream>>>(Bsel, Vld, Keep, out);
}
